// FusedMoE_26860725469445
// MI455X (gfx1250) — matrix operand order repaired and checked by reading, not compiled, not run on hardware
//
#include <hip/hip_runtime.h>
#include <math.h>

// ---------------- problem constants (match reference) ----------------
#define T_TOK 1024
#define HDIM  1024
#define IDIM  2816
#define NEXP  8
// top_k is 2 in the reference setup (d_in[6]); routing below hardcodes top-2.

#define MTILE 64
#define NTILE 64
#define KTILE 64               // two WMMA K-steps (2 x K=32) per staged tile
#define ROWCAP 2560            // 2*T + 8*63 rounded up to MTILE multiple

// ---------------- workspace layout (bytes) ----------------
#define OFF_XB    ((size_t)0)          // T*H bf16            = 2,097,152
#define OFF_TI    ((size_t)2097152)    // 2*T int             =     8,192
#define OFF_TW    ((size_t)2105344)    // 2*T float           =     8,192
#define OFF_OFFS  ((size_t)2113536)    // 9 ints (+pad)       =        64
#define OFF_RTOK  ((size_t)2113600)    // ROWCAP int          =    10,240
#define OFF_RW    ((size_t)2123840)    // ROWCAP float        =    10,240
#define OFF_ACT   ((size_t)2134080)    // ROWCAP*I bf16       = 14,417,920

typedef __attribute__((ext_vector_type(16))) __bf16 v16bf;
typedef __attribute__((ext_vector_type(8)))  float  v8f;

union FragAB { v16bf v; unsigned int u[8]; uint4 q[2]; };

// native f32 -> bf16 (fptrunc; backend lowers to the hardware converter)
static __device__ __forceinline__ unsigned short f2bf(float f) {
  union { __bf16 h; unsigned short u; } cv;
  cv.h = (__bf16)f;
  return cv.u;
}
static __device__ __forceinline__ unsigned int pack2bf(float a, float b) {
  union { __bf16 h[2]; unsigned int u; } cv;
  cv.h[0] = (__bf16)a;           // low 16 = even K element
  cv.h[1] = (__bf16)b;
  return cv.u;
}

// LDS tiles: [64 rows][32 dwords] = 64 bf16 of K per row. ks selects K-half (0/1).
// A-matrix 16x32 bf16 fragment (ISA 7.12.2): two 16B-aligned 4-dword runs.
static __device__ __forceinline__ FragAB loadA(const unsigned int* s, int rb, int ks, int lane) {
  int m = lane & 15, half = lane >> 4;
  int base = (rb * 16 + m) * 32 + ks * 16 + (half << 2);
  FragAB f;
  f.q[0] = *(const uint4*)&s[base];        // VGPRs 0-3: K pairs 0..7 (+half*8)
  f.q[1] = *(const uint4*)&s[base + 8];    // VGPRs 4-7: K pairs 16..23 (+half*8)
  return f;
}
// B-matrix 32x16 bf16 fragment, weight tile stored N-major: two 4-dword runs (same K order as A).
static __device__ __forceinline__ FragAB loadB(const unsigned int* s, int cb, int ks, int lane) {
  int n = lane & 15, half = lane >> 4;
  int base = (cb * 16 + n) * 32 + ks * 16 + (half << 2);
  FragAB f;
  f.q[0] = *(const uint4*)&s[base];
  f.q[1] = *(const uint4*)&s[base + 8];
  return f;
}
static __device__ __forceinline__ v8f wmma_bf16(FragAB a, FragAB b, v8f c) {
  return __builtin_amdgcn_wmma_f32_16x16x32_bf16(false, a.v, false, b.v, (short)0, c, false, false);
}

// 16 raw f32 (one thread's share of a 64x64 weight tile K-slice)
struct F16R { float4 f[4]; };
static __device__ __forceinline__ F16R ldB16(const float* p) {
  F16R r;
  r.f[0] = ((const float4*)p)[0];
  r.f[1] = ((const float4*)p)[1];
  r.f[2] = ((const float4*)p)[2];
  r.f[3] = ((const float4*)p)[3];
  return r;
}
static __device__ __forceinline__ void stB16(unsigned int* dst, const F16R& r) {
  ((uint4*)dst)[0] = make_uint4(pack2bf(r.f[0].x, r.f[0].y), pack2bf(r.f[0].z, r.f[0].w),
                                pack2bf(r.f[1].x, r.f[1].y), pack2bf(r.f[1].z, r.f[1].w));
  ((uint4*)dst)[1] = make_uint4(pack2bf(r.f[2].x, r.f[2].y), pack2bf(r.f[2].z, r.f[2].w),
                                pack2bf(r.f[3].x, r.f[3].y), pack2bf(r.f[3].z, r.f[3].w));
}

// ---------------- small utility kernels ----------------
__global__ void k_zero(float* p, int n) {
  int i = blockIdx.x * blockDim.x + threadIdx.x;
  if (i < n) p[i] = 0.0f;
}

__global__ void k_cvt_x(const float* __restrict__ x, unsigned int* __restrict__ xb, int n2) {
  int i = blockIdx.x * blockDim.x + threadIdx.x;   // one packed pair per thread
  if (i < n2) {
    float2 f = ((const float2*)x)[i];
    xb[i] = pack2bf(f.x, f.y);
  }
}

// softmax -> top-2 -> renormalize (per token)
__global__ void k_route(const float* __restrict__ logits, int* __restrict__ ti, float* __restrict__ tw) {
  int t = blockIdx.x * blockDim.x + threadIdx.x;
  if (t >= T_TOK) return;
  float p[NEXP];
  float mx = -1e30f;
#pragma unroll
  for (int e = 0; e < NEXP; ++e) { p[e] = logits[t * NEXP + e]; mx = fmaxf(mx, p[e]); }
#pragma unroll
  for (int e = 0; e < NEXP; ++e) p[e] = __expf(p[e] - mx);
  int i1 = 0; float v1 = p[0];
#pragma unroll
  for (int e = 1; e < NEXP; ++e) if (p[e] > v1) { v1 = p[e]; i1 = e; }
  int i2 = (i1 == 0) ? 1 : 0; float v2 = p[i2];
#pragma unroll
  for (int e = 0; e < NEXP; ++e) if (e != i1 && p[e] > v2) { v2 = p[e]; i2 = e; }
  float inv = 1.0f / (v1 + v2);      // softmax denom cancels in renormalization
  ti[t * 2 + 0] = i1; ti[t * 2 + 1] = i2;
  tw[t * 2 + 0] = v1 * inv; tw[t * 2 + 1] = v2 * inv;
}

// deterministic per-expert gather: count -> padded offsets -> ordered append
__global__ void k_gather(const int* __restrict__ ti, const float* __restrict__ tw,
                         int* __restrict__ offs, int* __restrict__ rowTok, float* __restrict__ rowW) {
  __shared__ int cnt[NEXP];
  __shared__ int off[NEXP + 1];
  int tid = threadIdx.x;
  if (tid < NEXP) {
    int c = 0;
    for (int t = 0; t < T_TOK; ++t) {
      if (ti[2 * t + 0] == tid) ++c;
      if (ti[2 * t + 1] == tid) ++c;
    }
    cnt[tid] = c;
  }
  __syncthreads();
  if (tid == 0) {
    off[0] = 0;
    for (int e = 0; e < NEXP; ++e) off[e + 1] = off[e] + ((cnt[e] + MTILE - 1) / MTILE) * MTILE;
    for (int e = 0; e <= NEXP; ++e) offs[e] = off[e];   // offs[8] == padded total rows
  }
  __syncthreads();
  if (tid < NEXP) {
    int pos = off[tid];
    for (int t = 0; t < T_TOK; ++t)
      for (int k = 0; k < 2; ++k)
        if (ti[2 * t + k] == tid) { rowTok[pos] = t; rowW[pos] = tw[2 * t + k]; ++pos; }
    for (; pos < off[tid + 1]; ++pos) { rowTok[pos] = -1; rowW[pos] = 0.0f; }
  }
}

// ---------------- GEMM1: h = x @ w13^T, SiLU-gate, act (bf16) ----------------
// Double-buffered LDS, one barrier per K-iteration, 8 WMMAs per iteration/wave.
__global__ __launch_bounds__(256) void k_gemm1(const unsigned short* __restrict__ xb,
                                               const float* __restrict__ w13,
                                               const float* __restrict__ s13,
                                               const int* __restrict__ offs,
                                               const int* __restrict__ rowTok,
                                               unsigned short* __restrict__ act) {
  __shared__ __align__(16) unsigned int sA[2][MTILE * 32];
  __shared__ __align__(16) unsigned int sBg[2][NTILE * 32];
  __shared__ __align__(16) unsigned int sBu[2][NTILE * 32];
  __shared__ int sTok[MTILE];

  int m0 = blockIdx.y * MTILE;
  if (m0 >= offs[NEXP]) return;
  int e = 0;
  while (offs[e + 1] <= m0) ++e;        // tiles never straddle experts (padded)
  int i0 = blockIdx.x * NTILE;
  const float* wg = w13 + (size_t)e * (2 * IDIM) * HDIM;

  int tid = threadIdx.x;
  if (tid < MTILE) sTok[tid] = rowTok[m0 + tid];
  __syncthreads();

  int lane = tid & 31, wave = tid >> 5;
  int cb = wave & 3, rb0 = (wave >> 2) * 2;
  v8f aG0 = {}, aG1 = {}, aU0 = {}, aU1 = {};

  int r = tid >> 2, q = tid & 3;
  int tok = sTok[r];
  const unsigned short* px = (tok >= 0) ? (xb + (size_t)tok * HDIM + q * 16) : nullptr;
  const float* pg = wg + (size_t)(i0 + r) * HDIM + q * 16;
  const float* pu = wg + (size_t)(IDIM + i0 + r) * HDIM + q * 16;
  unsigned int* stA = &sA[0][r * 32 + q * 8];   // buffer 1 at +MTILE*32
  unsigned int* stG = &sBg[0][r * 32 + q * 8];
  unsigned int* stU = &sBu[0][r * 32 + q * 8];

  const int NK = HDIM / KTILE;                  // 16

  // prologue: stage tile 0 into buffer 0
  {
    uint4 a0 = px ? ((const uint4*)px)[0] : make_uint4(0u, 0u, 0u, 0u);
    uint4 a1 = px ? ((const uint4*)px)[1] : make_uint4(0u, 0u, 0u, 0u);
    F16R g = ldB16(pg), u = ldB16(pu);
    ((uint4*)stA)[0] = a0; ((uint4*)stA)[1] = a1;
    stB16(stG, g);
    stB16(stU, u);
  }
  __syncthreads();

  for (int kt = 0; kt < NK; ++kt) {
    int cur = kt & 1;
    bool hasNext = (kt + 1) < NK;

    // issue next tile's global loads before computing (latency hiding)
    uint4 nA0 = make_uint4(0u, 0u, 0u, 0u), nA1 = nA0;
    F16R nG, nU;
    if (hasNext) {
      if (px) { nA0 = ((const uint4*)(px + (kt + 1) * KTILE))[0];
                nA1 = ((const uint4*)(px + (kt + 1) * KTILE))[1]; }
      nG = ldB16(pg + (kt + 1) * KTILE);
      nU = ldB16(pu + (kt + 1) * KTILE);
      if (kt + 2 < NK) {                        // global_prefetch_b8 two tiles ahead
        __builtin_prefetch(pg + (kt + 2) * KTILE, 0, 0);
        __builtin_prefetch(pu + (kt + 2) * KTILE, 0, 0);
      }
    }

    const unsigned int* bA = sA[cur];
    const unsigned int* bG = sBg[cur];
    const unsigned int* bU = sBu[cur];
#pragma unroll
    for (int ks = 0; ks < 2; ++ks) {
      FragAB a0 = loadA(bA, rb0, ks, lane);
      FragAB a1 = loadA(bA, rb0 + 1, ks, lane);
      FragAB bg = loadB(bG, cb, ks, lane);
      FragAB bu = loadB(bU, cb, ks, lane);
      aG0 = wmma_bf16(a0, bg, aG0);
      aG1 = wmma_bf16(a1, bg, aG1);
      aU0 = wmma_bf16(a0, bu, aU0);
      aU1 = wmma_bf16(a1, bu, aU1);
    }

    if (hasNext) {
      int nxt = cur ^ 1;
      unsigned int* dA = stA + (size_t)nxt * (MTILE * 32);
      unsigned int* dG = stG + (size_t)nxt * (NTILE * 32);
      unsigned int* dU = stU + (size_t)nxt * (NTILE * 32);
      ((uint4*)dA)[0] = nA0; ((uint4*)dA)[1] = nA1;
      stB16(dG, nG);
      stB16(dU, nU);
    }
    __syncthreads();
  }

  float sc = s13[e];
  int n = lane & 15, half = lane >> 4;
#pragma unroll
  for (int rbi = 0; rbi < 2; ++rbi) {
    v8f G = rbi ? aG1 : aG0;
    v8f U = rbi ? aU1 : aU0;
    int rb = rb0 + rbi;
#pragma unroll
    for (int d = 0; d < 8; ++d) {
      int m = rb * 16 + d + half * 8;
      if (sTok[m] >= 0) {
        float g = G[d] * sc, u = U[d] * sc;
        float a = (g / (1.0f + __expf(-g))) * u;     // silu(g)*u
        act[(size_t)(m0 + m) * IDIM + i0 + cb * 16 + n] = f2bf(a);
      }
    }
  }
}

// ---------------- GEMM2: out += combine * (act @ w2^T) ----------------
__global__ __launch_bounds__(256) void k_gemm2(const unsigned short* __restrict__ act,
                                               const float* __restrict__ w2,
                                               const float* __restrict__ s2,
                                               const int* __restrict__ offs,
                                               const int* __restrict__ rowTok,
                                               const float* __restrict__ rowW,
                                               float* __restrict__ out) {
  __shared__ __align__(16) unsigned int sA[2][MTILE * 32];
  __shared__ __align__(16) unsigned int sB[2][NTILE * 32];
  __shared__ int sTok[MTILE];
  __shared__ float sW[MTILE];

  int m0 = blockIdx.y * MTILE;
  if (m0 >= offs[NEXP]) return;
  int e = 0;
  while (offs[e + 1] <= m0) ++e;
  int h0 = blockIdx.x * NTILE;
  const float* wp = w2 + (size_t)e * HDIM * IDIM;

  int tid = threadIdx.x;
  if (tid < MTILE) { sTok[tid] = rowTok[m0 + tid]; sW[tid] = rowW[m0 + tid]; }
  __syncthreads();

  int lane = tid & 31, wave = tid >> 5;
  int cb = wave & 3, rb0 = (wave >> 2) * 2;
  v8f c0 = {}, c1 = {};

  int r = tid >> 2, q = tid & 3;
  const unsigned short* pa = act + (size_t)(m0 + r) * IDIM + q * 16;
  const float* pb = wp + (size_t)(h0 + r) * IDIM + q * 16;
  unsigned int* stA = &sA[0][r * 32 + q * 8];
  unsigned int* stB = &sB[0][r * 32 + q * 8];

  const int NK = IDIM / KTILE;                  // 44

  {
    uint4 a0 = ((const uint4*)pa)[0];
    uint4 a1 = ((const uint4*)pa)[1];
    F16R b = ldB16(pb);
    ((uint4*)stA)[0] = a0; ((uint4*)stA)[1] = a1;
    stB16(stB, b);
  }
  __syncthreads();

  for (int kt = 0; kt < NK; ++kt) {
    int cur = kt & 1;
    bool hasNext = (kt + 1) < NK;

    uint4 nA0 = make_uint4(0u, 0u, 0u, 0u), nA1 = nA0;
    F16R nB;
    if (hasNext) {
      nA0 = ((const uint4*)(pa + (kt + 1) * KTILE))[0];
      nA1 = ((const uint4*)(pa + (kt + 1) * KTILE))[1];
      nB  = ldB16(pb + (kt + 1) * KTILE);
      if (kt + 2 < NK) __builtin_prefetch(pb + (kt + 2) * KTILE, 0, 0);
    }

    const unsigned int* bA = sA[cur];
    const unsigned int* bB = sB[cur];
#pragma unroll
    for (int ks = 0; ks < 2; ++ks) {
      FragAB a0 = loadA(bA, rb0, ks, lane);
      FragAB a1 = loadA(bA, rb0 + 1, ks, lane);
      FragAB b  = loadB(bB, cb, ks, lane);
      c0 = wmma_bf16(a0, b, c0);
      c1 = wmma_bf16(a1, b, c1);
    }

    if (hasNext) {
      int nxt = cur ^ 1;
      unsigned int* dA = stA + (size_t)nxt * (MTILE * 32);
      unsigned int* dB = stB + (size_t)nxt * (NTILE * 32);
      ((uint4*)dA)[0] = nA0; ((uint4*)dA)[1] = nA1;
      stB16(dB, nB);
    }
    __syncthreads();
  }

  float sc = s2[e];
  int n = lane & 15, half = lane >> 4;
#pragma unroll
  for (int rbi = 0; rbi < 2; ++rbi) {
    v8f C = rbi ? c1 : c0;
    int rb = rb0 + rbi;
#pragma unroll
    for (int d = 0; d < 8; ++d) {
      int m = rb * 16 + d + half * 8;
      int t = sTok[m];
      if (t >= 0)
        atomicAdd(&out[(size_t)t * HDIM + h0 + cb * 16 + n], sW[m] * sc * C[d]);
    }
  }
}

// ---------------- launch ----------------
extern "C" void kernel_launch(void* const* d_in, const int* in_sizes, int n_in,
                              void* d_out, int out_size, void* d_ws, size_t ws_size,
                              hipStream_t stream) {
  (void)in_sizes; (void)n_in; (void)out_size; (void)ws_size;
  const float* x      = (const float*)d_in[0];
  const float* logits = (const float*)d_in[1];
  const float* w13    = (const float*)d_in[2];
  const float* w2     = (const float*)d_in[3];
  const float* s13    = (const float*)d_in[4];
  const float* s2     = (const float*)d_in[5];
  // d_in[6] = top_k (== 2; routing hardcodes top-2)
  float* out = (float*)d_out;
  char* ws = (char*)d_ws;

  unsigned short* xb  = (unsigned short*)(ws + OFF_XB);
  int*   ti           = (int*)(ws + OFF_TI);
  float* tw           = (float*)(ws + OFF_TW);
  int*   offs         = (int*)(ws + OFF_OFFS);
  int*   rowTok       = (int*)(ws + OFF_RTOK);
  float* rowW         = (float*)(ws + OFF_RW);
  unsigned short* act = (unsigned short*)(ws + OFF_ACT);

  int nOut = T_TOK * HDIM;
  k_zero<<<(nOut + 255) / 256, 256, 0, stream>>>(out, nOut);
  k_cvt_x<<<(nOut / 2 + 255) / 256, 256, 0, stream>>>(x, (unsigned int*)xb, nOut / 2);
  k_route<<<(T_TOK + 255) / 256, 256, 0, stream>>>(logits, ti, tw);
  k_gather<<<1, 64, 0, stream>>>(ti, tw, offs, rowTok, rowW);

  dim3 g1(IDIM / NTILE, ROWCAP / MTILE);   // 44 x 40
  k_gemm1<<<g1, 256, 0, stream>>>(xb, w13, s13, offs, rowTok, act);

  dim3 g2(HDIM / NTILE, ROWCAP / MTILE);   // 16 x 40
  k_gemm2<<<g2, 256, 0, stream>>>(act, w2, s2, offs, rowTok, rowW, out);
}
